// SpatialAttention_57157424775255
// MI455X (gfx1250) — matrix operand order repaired and checked by reading, not compiled, not run on hardware
//
#include <hip/hip_runtime.h>
#include <hip/hip_bf16.h>
#include <hip/hip_fp16.h>

typedef __attribute__((ext_vector_type(16))) _Float16 v16h;
typedef __attribute__((ext_vector_type(8)))  float    v8f;

#define B_   16
#define S_   128
#define L_   543
#define F_   128
#define H_   64
#define LPAD 544
#define NT   4      // N tiles  (H/16)
#define NK   4      // K steps  (F/32)
#define MT   34     // M tiles  (ceil(L/16))
#define NW   8      // waves per block

// Branchless tanh. gfx1250 has native V_TANH_F32; fall back to an
// exp/rcp form that saturates correctly at +/-inf without EXEC branches.
__device__ __forceinline__ float fast_tanh(float v) {
#if __has_builtin(__builtin_amdgcn_tanhf)
    return __builtin_amdgcn_tanhf(v);
#else
    float e = __expf(2.0f * v);                    // v_exp_f32 (trans)
    return 1.0f - 2.0f * __builtin_amdgcn_rcpf(e + 1.0f);  // v_rcp_f32
#endif
}

__global__ __launch_bounds__(256)
void spatial_attn_kernel(const float* __restrict__ x,
                         const float* __restrict__ W1,
                         const float* __restrict__ b1,
                         const float* __restrict__ W2,
                         float* __restrict__ out_att,   // (B*S, F)
                         float* __restrict__ out_p)     // (B*S, L)
{
    // W1 packed in exact v_wmma_f32_16x16x32_f16 B-fragment register order:
    // [nt][kk][lane][e] ; N=nt*16+lane%16, K=kk*32+(lane>=16?8:0)+(e&7)+(e>=8?16:0)
    __shared__ __align__(32) _Float16 bfrag[NT * NK * 32 * 16];   // 16 KB
    __shared__ float scores[LPAD];                                 // 2.1 KB
    __shared__ float b1s[H_], w2s[H_];
    __shared__ float wred[NW];
    __shared__ float bcast;
    __shared__ float pacc[8][F_];                                  // 4 KB

    const int tid  = threadIdx.x;
    const int lane = tid & 31;
    const int wid  = tid >> 5;
    const int g    = blockIdx.x;                    // b*S + s
    const size_t xbase = (size_t)g * (L_ * F_);
    const float4* x4 = (const float4*)(x + xbase);

    // ---- stage W1 (f16, B-fragment order), b1, W2 ----
    for (int i = tid; i < NT * NK * 32 * 16; i += 256) {
        int e  = i & 15;
        int ln = (i >> 4) & 31;
        int kk = (i >> 9) & 3;
        int nt = i >> 11;
        int n  = nt * 16 + (ln & 15);
        int k  = kk * 32 + ((ln >= 16) ? 8 : 0) + (e & 7) + ((e & 8) ? 16 : 0);
        bfrag[i] = (_Float16)W1[k * H_ + n];
    }
    if (tid < H_) { b1s[tid] = b1[tid]; w2s[tid] = W2[tid]; }
    __syncthreads();

    // ---- Phase 1: scores[l] = tanh(x@W1 + b1) @ W2 via WMMA ----
    const int mrow = lane & 15;
    const int hi   = lane >> 4;        // half-wave selects K sub-block per ISA A layout
    for (int tile = wid; tile < MT; tile += NW) {      // uniform per wave -> EXEC all-ones
        const int l0 = tile * 16;
        const int l  = l0 + mrow;
        // Row 543 (pad row of last tile) is clamped instead of zeroed: its score
        // lands in scores[543], which softmax (l < 543) never reads. This keeps
        // every load unconditional -> no EXEC-masked load blocks, full MLP.
        const int lc = (l < L_) ? l : (L_ - 1);
        const float* xrow = x + xbase + (size_t)lc * F_;

        // Issue all 16 b128 loads of this tile's A data, then convert.
        float4 raw[NK][4];
        #pragma unroll
        for (int kk = 0; kk < NK; ++kk) {
            const int k0 = kk * 32 + hi * 8;
            raw[kk][0] = *(const float4*)(xrow + k0     );
            raw[kk][1] = *(const float4*)(xrow + k0 +  4);
            raw[kk][2] = *(const float4*)(xrow + k0 + 16);
            raw[kk][3] = *(const float4*)(xrow + k0 + 20);
        }
        v16h afr[NK];
        #pragma unroll
        for (int kk = 0; kk < NK; ++kk) {
            #pragma unroll
            for (int q = 0; q < 4; ++q) {
                afr[kk][q * 4 + 0] = (_Float16)raw[kk][q].x;
                afr[kk][q * 4 + 1] = (_Float16)raw[kk][q].y;
                afr[kk][q * 4 + 2] = (_Float16)raw[kk][q].z;
                afr[kk][q * 4 + 3] = (_Float16)raw[kk][q].w;
            }
        }

        float part[8];
        #pragma unroll
        for (int j = 0; j < 8; ++j) part[j] = 0.f;

        #pragma unroll
        for (int nt = 0; nt < NT; ++nt) {
            v8f c = {};
            #pragma unroll
            for (int kk = 0; kk < NK; ++kk) {
                v16h bf = *(const v16h*)&bfrag[(((nt * NK + kk) * 32) + lane) * 16];
                c = __builtin_amdgcn_wmma_f32_16x16x32_f16(
                        false, afr[kk], false, bf, (short)0, c, false, false);
            }
            // fold tanh + W2 dot: lane holds (m = j + hi*8, n = nt*16 + lane%16)
            const int n    = nt * 16 + (lane & 15);
            const float wv = w2s[n];
            const float bv = b1s[n];
            #pragma unroll
            for (int j = 0; j < 8; ++j) part[j] += wv * fast_tanh(c[j] + bv);
        }

        // reduce over the 16 N-lanes of each half-wave -> per-landmark score
        #pragma unroll
        for (int j = 0; j < 8; ++j) {
            float v = part[j];
            v += __shfl_xor(v, 1, 32);
            v += __shfl_xor(v, 2, 32);
            v += __shfl_xor(v, 4, 32);
            v += __shfl_xor(v, 8, 32);
            if (lane == 0)  scores[l0 + j]     = v;   // rows 0..7  of tile
            if (lane == 16) scores[l0 + 8 + j] = v;   // rows 8..15 of tile
        }
    }
    __syncthreads();

    // ---- softmax over L (b2 is a constant shift: cancels in softmax) ----
    float lmax = -3.402823466e38f;
    for (int i = tid; i < L_; i += 256) lmax = fmaxf(lmax, scores[i]);
    for (int m = 16; m >= 1; m >>= 1) lmax = fmaxf(lmax, __shfl_xor(lmax, m, 32));
    if (lane == 0) wred[wid] = lmax;
    __syncthreads();
    if (tid == 0) {
        float m = wred[0];
        for (int i = 1; i < NW; ++i) m = fmaxf(m, wred[i]);
        bcast = m;
    }
    __syncthreads();
    const float gmax = bcast;

    float lsum = 0.f;
    for (int i = tid; i < L_; i += 256) {
        float e = __expf(scores[i] - gmax);
        scores[i] = e;
        lsum += e;
    }
    for (int m = 16; m >= 1; m >>= 1) lsum += __shfl_xor(lsum, m, 32);
    if (lane == 0) wred[wid] = lsum;
    __syncthreads();
    if (tid == 0) {
        float s = 0.f;
        for (int i = 0; i < NW; ++i) s += wred[i];
        bcast = 1.0f / s;
    }
    __syncthreads();
    const float inv = bcast;
    for (int i = tid; i < L_; i += 256) {
        float a = scores[i] * inv;
        scores[i] = a;
        out_p[(size_t)g * L_ + i] = a;          // attn output
    }
    __syncthreads();

    // ---- Phase 2: attended[f] = sum_l attn[l] * x[l][f]  (x slice now L2-hot) ----
    const int cg = tid & 31;     // float4 column group (covers F=128)
    const int rg = tid >> 5;     // row-stride group 0..7
    float4 acc = make_float4(0.f, 0.f, 0.f, 0.f);
    for (int l2 = rg; l2 < L_; l2 += 8) {
        float4 v = x4[l2 * 32 + cg];
        float a = scores[l2];
        acc.x += a * v.x; acc.y += a * v.y; acc.z += a * v.z; acc.w += a * v.w;
    }
    pacc[rg][cg * 4 + 0] = acc.x;
    pacc[rg][cg * 4 + 1] = acc.y;
    pacc[rg][cg * 4 + 2] = acc.z;
    pacc[rg][cg * 4 + 3] = acc.w;
    __syncthreads();
    if (tid < F_) {
        float s = 0.f;
        #pragma unroll
        for (int r = 0; r < 8; ++r) s += pacc[r][tid];
        out_att[(size_t)g * F_ + tid] = s;
    }
}

extern "C" void kernel_launch(void* const* d_in, const int* in_sizes, int n_in,
                              void* d_out, int out_size, void* d_ws, size_t ws_size,
                              hipStream_t stream) {
    const float* x  = (const float*)d_in[0];
    const float* W1 = (const float*)d_in[1];
    const float* b1 = (const float*)d_in[2];
    const float* W2 = (const float*)d_in[3];
    // d_in[4] = b2: softmax(scores + b2) == softmax(scores); provably unused.
    float* out_att = (float*)d_out;                         // (B*S*F) attended
    float* out_p   = (float*)d_out + (size_t)B_ * S_ * F_;  // (B*S*L) attn
    dim3 grid(B_ * S_), block(256);
    hipLaunchKernelGGL(spatial_attn_kernel, grid, block, 0, stream,
                       x, W1, b1, W2, out_att, out_p);
}
